// ModernNCA_14319420965291
// MI455X (gfx1250) — matrix operand order repaired and checked by reading, not compiled, not run on hardware
//
#include <hip/hip_runtime.h>
#include <hip/hip_bf16.h>
#include <stdint.h>

// ---------------------------------------------------------------------------
// ModernNCA forward, fused for MI455X (gfx1250, wave32, WMMA).
//   K1 hist      : per-256-block class histograms (deterministic)
//   K2 scan      : padded class offsets (256-aligned) + per-block bases
//   K3 scatter   : deterministic counting-sort permutation of candidates
//   K4 encode(x) : z = xW^T+b via bf16x3 WMMA -> zh/zl bf16 (true K order) + ||z||^2
//   K5 encode(cx): permuted zc encode -> ch/cl bf16 (true K order) + ||zc||^2
//   K6 distance  : 128x256 C-tiles, bf16x3 WMMA dot, exp(-sqrt(..)),
//                  deterministic per-row block sums -> Partial[mblock][n]
//   K7 finalize  : out[n][c] = log(sum_c / sum_all + 1e-7)
//
// K order: the A-operand WMMA layout reads K positions
// {8h..8h+7, 16+8h..16+8h+7} per 32-chunk; no [8,16)<->[16,24) swap is applied:
// Zh/Zl and Ch/Cl stay in true K order, and each A fragment is one contiguous 32B load
// (k = 16h..16h+15), as on the B side.  Dot products are K-permutation invariant
// as long as A and B use the same k order, which both sides do here.
// ---------------------------------------------------------------------------

typedef __attribute__((ext_vector_type(16))) __bf16 v16bf;
typedef __attribute__((ext_vector_type(8)))  float  v8f;

#define DOUT 10
#define DIM  128

static __device__ __forceinline__ v8f wmma_bf16(v16bf a, v16bf b, v8f c) {
  return __builtin_amdgcn_wmma_f32_16x16x32_bf16(false, a, false, b, (short)0, c,
                                                 false, false);
}

// involution on K%32: swap [8,16) <-> [16,24)
static __device__ __forceinline__ int kswap(int col) {
  int i = col & 31, g = (i >> 3) & 3;
  int j = (g == 1) ? i + 8 : (g == 2) ? i - 8 : i;
  return (col & ~31) | j;
}

// ---------------- K1: per-block class histogram ----------------------------
__global__ void __launch_bounds__(256) nca_hist(const int* __restrict__ y, int M,
                                                int* __restrict__ blockHist) {
  __shared__ int cnt[DOUT];
  if (threadIdx.x < DOUT) cnt[threadIdx.x] = 0;
  __syncthreads();
  int m = blockIdx.x * 256 + threadIdx.x;
  if (m < M) atomicAdd(&cnt[y[m]], 1);
  __syncthreads();
  if (threadIdx.x < DOUT)
    blockHist[(size_t)blockIdx.x * DOUT + threadIdx.x] = cnt[threadIdx.x];
}

// ---------------- K2: totals -> padded offsets + per-block bases -----------
__global__ void __launch_bounds__(256) nca_scan(const int* __restrict__ blockHist,
                                                int nBlocks,
                                                int* __restrict__ padOff,
                                                int* __restrict__ blockBase) {
  __shared__ int red[256];
  __shared__ int offs[DOUT + 1];
  const int t = threadIdx.x;
  for (int c = 0; c < DOUT; ++c) {
    int s = 0;
    for (int b = t; b < nBlocks; b += 256) s += blockHist[(size_t)b * DOUT + c];
    red[t] = s;
    __syncthreads();
    for (int st = 128; st > 0; st >>= 1) {
      if (t < st) red[t] += red[t + st];
      __syncthreads();
    }
    if (t == 0) offs[c] = red[0];
    __syncthreads();
  }
  if (t == 0) {
    int o = 0;
    for (int c = 0; c < DOUT; ++c) {
      int tot = offs[c];
      offs[c] = o;
      o += ((tot + 255) >> 8) << 8;   // pad class segment to 256
    }
    offs[DOUT] = o;
    for (int c = 0; c <= DOUT; ++c) padOff[c] = offs[c];
  }
  __syncthreads();
  if (t < DOUT) {
    int cur = offs[t];
#pragma unroll 8
    for (int b = 0; b < nBlocks; ++b) {
      blockBase[(size_t)b * DOUT + t] = cur;
      cur += blockHist[(size_t)b * DOUT + t];
    }
  }
}

// ---------------- K3: deterministic scatter (counting sort) ----------------
__global__ void __launch_bounds__(256) nca_scatter(const int* __restrict__ y, int M,
                                                   const int* __restrict__ blockBase,
                                                   int* __restrict__ perm) {
  __shared__ int cls[256];
  const int tid = threadIdx.x;
  const int m = blockIdx.x * 256 + tid;
  int c = (m < M) ? y[m] : -1;
  cls[tid] = c;
  __syncthreads();
  if (m < M) {
    int rank = 0;
    for (int u = 0; u < tid; ++u) rank += (cls[u] == c);
    perm[blockBase[(size_t)blockIdx.x * DOUT + c] + rank] = m;
  }
}

// ---------------- K4/K5: encoder (bf16x3 WMMA, dual accumulators) ----------
// Block = 128 threads (4 waves); each wave encodes 16 rows x 128 features.
__global__ void __launch_bounds__(128) nca_encode(
    const float* __restrict__ X, const float* __restrict__ W,
    const float* __restrict__ bias, const int* __restrict__ perm,
    __bf16* __restrict__ Zh, __bf16* __restrict__ Zl,
    float* __restrict__ Nrm, int permuted, int kswz) {
  const int lane = threadIdx.x & 31;
  const int wv   = threadIdx.x >> 5;
  const int h    = lane >> 4;       // wave half
  const int c16  = lane & 15;
  const int r0   = blockIdx.x * 64 + wv * 16;

  // --- A fragments: this lane supplies row (r0+c16) of X -------------------
  int srcA = permuted ? perm[r0 + c16] : (r0 + c16);
  const float* arow = X + (size_t)(srcA < 0 ? 0 : srcA) * DIM;
  const float av = (srcA >= 0) ? 1.0f : 0.0f;

  v16bf Ah[4], Al[4];
#pragma unroll
  for (int kc = 0; kc < 4; ++kc) {
    const int kb = kc * 32;
    const float4* p0 = (const float4*)(arow + kb + 16 * h);
    const float4* p1 = (const float4*)(arow + kb + 16 * h + 8);
    float4 a0 = p0[0], a1 = p0[1], b0 = p1[0], b1 = p1[1];
    float f[16] = {a0.x, a0.y, a0.z, a0.w, a1.x, a1.y, a1.z, a1.w,
                   b0.x, b0.y, b0.z, b0.w, b1.x, b1.y, b1.z, b1.w};
#pragma unroll
    for (int e = 0; e < 16; ++e) {
      float v = f[e] * av;
      __bf16 hi = (__bf16)v;
      Ah[kc][e] = hi;
      Al[kc][e] = (__bf16)(v - (float)hi);
    }
  }

  // pad flags for this lane's C rows (rows r0+8h+j)
  int pad[8];
#pragma unroll
  for (int j = 0; j < 8; ++j) {
    int row = r0 + 8 * h + j;
    int s = permuted ? perm[row] : row;
    pad[j] = (s < 0);
  }

  float nrm[8] = {0.f, 0.f, 0.f, 0.f, 0.f, 0.f, 0.f, 0.f};

  for (int ft = 0; ft < 8; ft += 2) {       // feature tiles, two at a time
    v8f acc0 = {0.f, 0.f, 0.f, 0.f, 0.f, 0.f, 0.f, 0.f};
    v8f acc1 = {0.f, 0.f, 0.f, 0.f, 0.f, 0.f, 0.f, 0.f};
#pragma unroll
    for (int kc = 0; kc < 4; ++kc) {
      v16bf B0h, B0l, B1h, B1l;
#pragma unroll
      for (int u = 0; u < 2; ++u) {
        const float* wrow =
            W + (size_t)((ft + u) * 16 + c16) * DIM + kc * 32 + 16 * h;
        const float4* wp = (const float4*)wrow;
        float4 w0 = wp[0], w1 = wp[1], w2 = wp[2], w3 = wp[3];
        float f[16] = {w0.x, w0.y, w0.z, w0.w, w1.x, w1.y, w1.z, w1.w,
                       w2.x, w2.y, w2.z, w2.w, w3.x, w3.y, w3.z, w3.w};
#pragma unroll
        for (int e = 0; e < 16; ++e) {
          __bf16 hi = (__bf16)f[e];
          __bf16 lo = (__bf16)(f[e] - (float)hi);
          if (u == 0) { B0h[e] = hi; B0l[e] = lo; }
          else        { B1h[e] = hi; B1l[e] = lo; }
        }
      }
      // interleave the two accumulator chains (dep distance 2)
      acc0 = wmma_bf16(Al[kc], B0h, acc0);
      acc1 = wmma_bf16(Al[kc], B1h, acc1);
      acc0 = wmma_bf16(Ah[kc], B0l, acc0);
      acc1 = wmma_bf16(Ah[kc], B1l, acc1);
      acc0 = wmma_bf16(Ah[kc], B0h, acc0);
      acc1 = wmma_bf16(Ah[kc], B1h, acc1);
    }
#pragma unroll
    for (int u = 0; u < 2; ++u) {
      const int col = (ft + u) * 16 + c16;
      const float bv = bias[col];
      const int colS = col;   // true K order
#pragma unroll
      for (int j = 0; j < 8; ++j) {
        float z = (u == 0 ? acc0[j] : acc1[j]) + bv;
        if (pad[j]) z = 0.0f;
        __bf16 hi = (__bf16)z;
        size_t idx = (size_t)(r0 + 8 * h + j) * DIM + colS;
        Zh[idx] = hi;
        Zl[idx] = (__bf16)(z - (float)hi);
        nrm[j] += z * z;
      }
    }
  }

  // reduce squared norms across the 16 lanes of this half
#pragma unroll
  for (int j = 0; j < 8; ++j) {
    float s = nrm[j];
    s += __shfl_xor(s, 1, 32);
    s += __shfl_xor(s, 2, 32);
    s += __shfl_xor(s, 4, 32);
    s += __shfl_xor(s, 8, 32);
    nrm[j] = s;
  }
  if (c16 == 0) {
#pragma unroll
    for (int j = 0; j < 8; ++j)
      Nrm[r0 + 8 * h + j] = pad[j] ? 1e30f : nrm[j];
  }
}

// ---------------- K6: distance + exp + per-row block sums ------------------
// Block = 256 threads (8 waves = 2 wave-rows x 4 wave-cols), C-tile 128x256.
__global__ void __launch_bounds__(256) nca_dist(
    const __bf16* __restrict__ Zh, const __bf16* __restrict__ Zl,
    const float* __restrict__ Q,
    const __bf16* __restrict__ Ch, const __bf16* __restrict__ Cl,
    const float* __restrict__ R,
    float* __restrict__ Partial, int Nrows) {
  __shared__ float S[128][4];
  const int lane = threadIdx.x & 31;
  const int wv   = threadIdx.x >> 5;
  const int wr   = wv >> 2, wc = wv & 3;
  const int h    = lane >> 4, c16 = lane & 15;
  const int n0   = blockIdx.x * 128;
  const int nw0  = n0 + wr * 64;
  const int mw0  = blockIdx.y * 256 + wc * 64;

  v8f acc[4][4];
#pragma unroll
  for (int a = 0; a < 4; ++a)
#pragma unroll
    for (int b = 0; b < 4; ++b)
      acc[a][b] = (v8f){0.f, 0.f, 0.f, 0.f, 0.f, 0.f, 0.f, 0.f};

#pragma unroll
  for (int kc = 0; kc < 4; ++kc) {
    const int kb = kc * 32;
    v16bf Ahf[4], Alf[4], Bhf[4], Blf[4];
    // A fragments: one contiguous 32B load (true K order storage)
#pragma unroll
    for (int nt = 0; nt < 4; ++nt) {
      size_t base = (size_t)(nw0 + nt * 16 + c16) * DIM + kb + 16 * h;
      Ahf[nt] = *(const v16bf*)(Zh + base);
      Alf[nt] = *(const v16bf*)(Zl + base);
    }
    // B fragments: contiguous by construction (true K order)
#pragma unroll
    for (int mt = 0; mt < 4; ++mt) {
      size_t base = (size_t)(mw0 + mt * 16 + c16) * DIM + kb + 16 * h;
      Bhf[mt] = *(const v16bf*)(Ch + base);
      Blf[mt] = *(const v16bf*)(Cl + base);
    }
    // term-major ordering: consecutive WMMAs hit different accumulators
#pragma unroll
    for (int nt = 0; nt < 4; ++nt)
#pragma unroll
      for (int mt = 0; mt < 4; ++mt)
        acc[nt][mt] = wmma_bf16(Alf[nt], Bhf[mt], acc[nt][mt]);
#pragma unroll
    for (int nt = 0; nt < 4; ++nt)
#pragma unroll
      for (int mt = 0; mt < 4; ++mt)
        acc[nt][mt] = wmma_bf16(Ahf[nt], Blf[mt], acc[nt][mt]);
#pragma unroll
    for (int nt = 0; nt < 4; ++nt)
#pragma unroll
      for (int mt = 0; mt < 4; ++mt)
        acc[nt][mt] = wmma_bf16(Ahf[nt], Bhf[mt], acc[nt][mt]);
  }

  float rm[4];
#pragma unroll
  for (int mt = 0; mt < 4; ++mt) rm[mt] = R[mw0 + mt * 16 + c16];

  float rowsum[4][8];
#pragma unroll
  for (int nt = 0; nt < 4; ++nt)
#pragma unroll
    for (int j = 0; j < 8; ++j) rowsum[nt][j] = 0.f;

#pragma unroll
  for (int nt = 0; nt < 4; ++nt) {
    const float4* qp = (const float4*)(Q + nw0 + nt * 16 + 8 * h);
    float4 qa = qp[0], qb = qp[1];
    float q[8] = {qa.x, qa.y, qa.z, qa.w, qb.x, qb.y, qb.z, qb.w};
#pragma unroll
    for (int mt = 0; mt < 4; ++mt)
#pragma unroll
      for (int j = 0; j < 8; ++j) {
        float sq = q[j] + rm[mt] - 2.0f * acc[nt][mt][j];
        sq = fmaxf(sq, 1e-12f);
        rowsum[nt][j] += __expf(-sqrtf(sq));   // TEMP = 1
      }
  }

  // reduce over the 16 columns of each half, stash per (row, wave-col)
#pragma unroll
  for (int nt = 0; nt < 4; ++nt)
#pragma unroll
    for (int j = 0; j < 8; ++j) {
      float s = rowsum[nt][j];
      s += __shfl_xor(s, 1, 32);
      s += __shfl_xor(s, 2, 32);
      s += __shfl_xor(s, 4, 32);
      s += __shfl_xor(s, 8, 32);
      if (c16 == 0) S[wr * 64 + nt * 16 + 8 * h + j][wc] = s;
    }
  __syncthreads();
  if (threadIdx.x < 128) {
    float v = S[threadIdx.x][0] + S[threadIdx.x][1] + S[threadIdx.x][2] +
              S[threadIdx.x][3];
    Partial[(size_t)blockIdx.y * Nrows + n0 + threadIdx.x] = v;
  }
}

// ---------------- K7: finalize ---------------------------------------------
__global__ void __launch_bounds__(256) nca_finalize(
    const float* __restrict__ Partial, const int* __restrict__ padOff,
    float* __restrict__ Out, int Nrows, int numMB) {
  int n = blockIdx.x * blockDim.x + threadIdx.x;
  if (n >= Nrows) return;
  float tot = 0.f;
  for (int mb = 0; mb < numMB; ++mb) tot += Partial[(size_t)mb * Nrows + n];
  float inv = 1.0f / fmaxf(tot, 1e-30f);
  int off[DOUT + 1];
#pragma unroll
  for (int c = 0; c <= DOUT; ++c) off[c] = padOff[c];
#pragma unroll
  for (int c = 0; c < DOUT; ++c) {
    float s = 0.f;
    for (int mb = off[c] >> 8; mb < (off[c + 1] >> 8); ++mb)
      s += Partial[(size_t)mb * Nrows + n];
    Out[n * DOUT + c] = __logf(s * inv + 1e-7f);
  }
}

// ---------------------------------------------------------------------------
extern "C" void kernel_launch(void* const* d_in, const int* in_sizes, int n_in,
                              void* d_out, int out_size, void* d_ws, size_t ws_size,
                              hipStream_t stream) {
  const float* x  = (const float*)d_in[0];
  const float* cx = (const float*)d_in[1];
  const int*   cy = (const int*)d_in[2];
  const float* W  = (const float*)d_in[3];
  const float* b  = (const float*)d_in[4];

  const int N = in_sizes[0] / DIM;                 // 2048
  const int M = in_sizes[2];                       // 131072
  const int nBlocks = (M + 255) / 256;             // 512
  const int MpadMax = nBlocks * 256 + DOUT * 256;  // 133632 (worst-case padded)
  const int numMB = MpadMax / 256;                 // 522

  uint8_t* base = (uint8_t*)d_ws;
  size_t off = 0;
  auto carve = [&](size_t bytes) -> void* {
    void* p = base + off;
    off += (bytes + 255) & ~(size_t)255;
    return p;
  };
  int*    padOff    = (int*)carve(sizeof(int) * (DOUT + 1));
  int*    blockHist = (int*)carve(sizeof(int) * (size_t)nBlocks * DOUT);
  int*    blockBase = (int*)carve(sizeof(int) * (size_t)nBlocks * DOUT);
  int*    perm      = (int*)carve(sizeof(int) * (size_t)MpadMax);
  __bf16* Zh        = (__bf16*)carve(sizeof(__bf16) * (size_t)N * DIM);
  __bf16* Zl        = (__bf16*)carve(sizeof(__bf16) * (size_t)N * DIM);
  float*  Q         = (float*)carve(sizeof(float) * (size_t)N);
  __bf16* Ch        = (__bf16*)carve(sizeof(__bf16) * (size_t)MpadMax * DIM);
  __bf16* Cl        = (__bf16*)carve(sizeof(__bf16) * (size_t)MpadMax * DIM);
  float*  R         = (float*)carve(sizeof(float) * (size_t)MpadMax);
  float*  Partial   = (float*)carve(sizeof(float) * (size_t)numMB * N);

  // perm = -1 everywhere (unfilled padded slots -> pad rows)
  hipMemsetAsync(perm, 0xFF, sizeof(int) * (size_t)MpadMax, stream);

  nca_hist<<<nBlocks, 256, 0, stream>>>(cy, M, blockHist);
  nca_scan<<<1, 256, 0, stream>>>(blockHist, nBlocks, padOff, blockBase);
  nca_scatter<<<nBlocks, 256, 0, stream>>>(cy, M, blockBase, perm);

  // queries (A side) and candidates (B side) both stored in true K order
  nca_encode<<<N / 64, 128, 0, stream>>>(x, W, b, perm, Zh, Zl, Q, 0, 1);
  nca_encode<<<MpadMax / 64, 128, 0, stream>>>(cx, W, b, perm, Ch, Cl, R, 1, 0);

  dim3 grid(N / 128, numMB);
  nca_dist<<<grid, 256, 0, stream>>>(Zh, Zl, Q, Ch, Cl, R, Partial, N);

  nca_finalize<<<(N + 255) / 256, 256, 0, stream>>>(Partial, padOff,
                                                    (float*)d_out, N, numMB);
}
